// MambaSequenceEncoder_22299470201655
// MI455X (gfx1250) — matrix operand order repaired and checked by reading, compiled, not run on hardware
//
#include <hip/hip_runtime.h>
#include <hip/hip_bf16.h>
#include <math.h>

typedef __attribute__((ext_vector_type(16))) _Float16 v16h;
typedef __attribute__((ext_vector_type(8)))  _Float16 v8h;
typedef __attribute__((ext_vector_type(8)))  float    v8f;
typedef __attribute__((ext_vector_type(2)))  __fp16   v2fp16;

#define BB     8
#define LL     1024
#define DM     256
#define DI     512
#define DSTATE 16
#define MROWS  (BB * LL)   // 8192 token rows

// ---------------------------------------------------------------------------
// Weight pre-pack: f32 row-major W[K][N] -> f16 wave32 WMMA B-fragments.
// Packed index = ((nt*KT + kt)*32 + lane)*16 + e,
// element e of lane:  K = kt*32 + 16*(lane>>4) + e,  N = nt*16 + (lane&15)
// ---------------------------------------------------------------------------
__global__ void pack_w_f16(const float* __restrict__ W, _Float16* __restrict__ dst,
                           int K, int N) {
    int tid = blockIdx.x * blockDim.x + threadIdx.x;
    if (tid >= K * N) return;
    int e   = tid & 15;
    int ln  = (tid >> 4) & 31;
    int t   = tid >> 9;           // nt*KT + kt
    int KT  = K >> 5;
    int nt  = t / KT;
    int kt  = t % KT;
    int k   = kt * 32 + ((ln >> 4) << 4) + e;
    int n   = nt * 16 + (ln & 15);
    dst[tid] = (_Float16)W[(size_t)k * N + n];
}

// ---------------------------------------------------------------------------
// Block-wide sum reduction over 256 threads
// ---------------------------------------------------------------------------
__device__ __forceinline__ float block_sum(float v, float* red) {
    int tid = threadIdx.x;
    red[tid] = v;
    __syncthreads();
    for (int s = 128; s > 0; s >>= 1) {
        if (tid < s) red[tid] += red[tid + s];
        __syncthreads();
    }
    float r = red[0];
    __syncthreads();
    return r;
}

// input projection (pointwise conv over 4 channels) + layernorm -> h [M,256]
__global__ void input_proj_ln(const float* __restrict__ x, const float* __restrict__ w_in,
                              const float* __restrict__ b_in, const float* __restrict__ g,
                              const float* __restrict__ bln, float* __restrict__ h) {
    __shared__ float red[256];
    int row = blockIdx.x;              // b*L + l
    int b = row >> 10, l = row & 1023;
    int d = threadIdx.x;
    float v = b_in[d];
#pragma unroll
    for (int c = 0; c < 4; ++c)
        v += x[((size_t)(b * 4 + c)) * LL + l] * w_in[d * 4 + c];
    float m  = block_sum(v, red) * (1.0f / DM);
    float dv = v - m;
    float va = block_sum(dv * dv, red) * (1.0f / DM);
    h[(size_t)row * DM + d] = dv * rsqrtf(va + 1e-5f) * g[d] + bln[d];
}

// row layernorm over d_model
__global__ void ln_rows(const float* __restrict__ hin, const float* __restrict__ g,
                        const float* __restrict__ bln, float* __restrict__ hout) {
    __shared__ float red[256];
    int row = blockIdx.x;
    int d = threadIdx.x;
    float v  = hin[(size_t)row * DM + d];
    float m  = block_sum(v, red) * (1.0f / DM);
    float dv = v - m;
    float va = block_sum(dv * dv, red) * (1.0f / DM);
    hout[(size_t)row * DM + d] = dv * rsqrtf(va + 1e-5f) * g[d] + bln[d];
}

// ---------------------------------------------------------------------------
// Stage one 128x32 f32 tile -> f16 LDS (stride 40 halves) using packed
// converts: float2 load -> v_cvt_pk_rtz_f16_f32 -> one ds_store_b32.
// ---------------------------------------------------------------------------
template <int K>
__device__ __forceinline__ void stage_tile(const float* __restrict__ A, int m0,
                                           int kcol, _Float16* __restrict__ dst,
                                           int tid) {
#pragma unroll
    for (int j = 0; j < 8; ++j) {
        int flat = j * 256 + tid;          // pair index over 128x16 pairs
        int r  = flat >> 4;
        int cp = flat & 15;
        const float2 f = *(const float2*)&A[(size_t)(m0 + r) * K + kcol + cp * 2];
        v2fp16 p = __builtin_amdgcn_cvt_pkrtz(f.x, f.y);
        *(v2fp16*)&dst[r * 40 + cp * 2] = p;
    }
}

// ---------------------------------------------------------------------------
// WMMA GEMM: out[M,N] (+epilogue) = A[M,K] (f32 -> f16 via LDS, double buffered)
//            @ W (pre-packed f16 fragments).  M=8192; K,N compile-time.
// Block: 256 threads = 8 waves, tile 128(M) x 32(N); wave = 2 Mtiles x 1 Ntile.
// EPI: 0 = store, 1 = softplus(acc + bias[n]), 2 = residual add into out.
// ---------------------------------------------------------------------------
template <int K, int N, int EPI>
__global__ void gemm_wmma(const float* __restrict__ A, const _Float16* __restrict__ Bpk,
                          float* __restrict__ out, const float* __restrict__ bias) {
    constexpr int AS = 40;        // LDS row stride (halves), keeps 16B alignment
    constexpr int KT = K / 32;
    __shared__ __align__(16) _Float16 Ash[2][128 * AS];

    const int tid = threadIdx.x;
    const int w   = tid >> 5;
    const int ln  = tid & 31;
    const int mw  = w & 3;          // 4 M-wave rows of 32
    const int nw  = w >> 2;         // 2 N-wave cols of 16
    const int m0  = blockIdx.x * 128;
    const int ntg = blockIdx.y * 2 + nw;
    const int g   = ln >> 4;
    const int lr  = ln & 15;

    v8f acc0 = {};
    v8f acc1 = {};

    stage_tile<K>(A, m0, 0, Ash[0], tid);
    __syncthreads();

    for (int kt = 0; kt < KT; ++kt) {
        const _Float16* cur = Ash[kt & 1];
        _Float16*       nxt = Ash[(kt + 1) & 1];

        // A fragments from LDS: per lane two contiguous 8-half groups
        v16h a0, a1;
        {
            const _Float16* base = &cur[(mw * 32 + lr) * AS + g * 16];
            v8h lo = *(const v8h*)base;
            v8h hi = *(const v8h*)(base + 8);
#pragma unroll
            for (int j = 0; j < 8; ++j) { a0[j] = lo[j]; a0[8 + j] = hi[j]; }
        }
        {
            const _Float16* base = &cur[(mw * 32 + 16 + lr) * AS + g * 16];
            v8h lo = *(const v8h*)base;
            v8h hi = *(const v8h*)(base + 8);
#pragma unroll
            for (int j = 0; j < 8; ++j) { a1[j] = lo[j]; a1[8 + j] = hi[j]; }
        }

        // B fragment: 16 contiguous halves per lane (32B)
        v16h bf = *(const v16h*)(Bpk + ((size_t)(ntg * KT + kt) * 32 + ln) * 16);

        // prefetch next A tile into the other LDS buffer (overlaps WMMA)
        if (kt + 1 < KT)
            stage_tile<K>(A, m0, (kt + 1) * 32, nxt, tid);

        acc0 = __builtin_amdgcn_wmma_f32_16x16x32_f16(false, a0, false, bf,
                                                      (short)0, acc0, false, false);
        acc1 = __builtin_amdgcn_wmma_f32_16x16x32_f16(false, a1, false, bf,
                                                      (short)0, acc1, false, false);
        __syncthreads();
    }

    // epilogue: D layout -> element (vgpr i, lane) = row 8*g+i, col lr
    (void)bias;
    const int n = ntg * 16 + lr;
    float* p0 = out + (size_t)(m0 + mw * 32 + g * 8) * N + n;
    float* p1 = p0 + (size_t)16 * N;
#pragma unroll
    for (int i = 0; i < 8; ++i) {
        float v0 = acc0[i], v1 = acc1[i];
        if constexpr (EPI == 1) {
            float bz = bias[n];
            v0 += bz; v1 += bz;
            v0 = fmaxf(v0, 0.f) + log1pf(expf(-fabsf(v0)));
            v1 = fmaxf(v1, 0.f) + log1pf(expf(-fabsf(v1)));
            p0[(size_t)i * N] = v0;
            p1[(size_t)i * N] = v1;
        } else if constexpr (EPI == 2) {
            p0[(size_t)i * N] += v0;
            p1[(size_t)i * N] += v1;
        } else {
            p0[(size_t)i * N] = v0;
            p1[(size_t)i * N] = v1;
        }
    }
}

// causal depthwise conv (k=4, left pad 3) + bias + SiLU over xz[:, :DI]
__global__ void conv_silu(const float* __restrict__ xz, const float* __restrict__ cw,
                          const float* __restrict__ cb, float* __restrict__ xo) {
    int idx = blockIdx.x * blockDim.x + threadIdx.x;   // M * DI
    int d   = idx & (DI - 1);
    int row = idx >> 9;                                // b*L + l
    int l   = row & (LL - 1);
    float acc = cb[d];
#pragma unroll
    for (int j = 0; j < 4; ++j) {
        int ll = l + j - 3;
        if (ll >= 0)
            acc += xz[(size_t)(row + j - 3) * (2 * DI) + d] * cw[d * 4 + j];
    }
    xo[(size_t)row * DI + d] = acc / (1.f + expf(-acc));
}

// selective scan: block = (batch, 16-channel chunk); thread = (channel, state)
// fuses +D*x and *silu(z); writes yz [M, DI]
__global__ void ssm_scan(const float* __restrict__ xin, const float* __restrict__ dt,
                         const float* __restrict__ bc, const float* __restrict__ xz,
                         const float* __restrict__ A_log, const float* __restrict__ Dp,
                         float* __restrict__ yz) {
    int b  = blockIdx.x >> 5;
    int dc = blockIdx.x & 31;
    int tid = threadIdx.x;
    int dl = tid >> 4, s = tid & 15;
    int d  = dc * 16 + dl;

    float a   = -expf(A_log[d * DSTATE + s]);
    float Dd  = Dp[d];
    float hst = 0.f;
    size_t base = (size_t)b * LL;

    for (int t = 0; t < LL; ++t) {
        size_t row = base + t;
        float xt  = xin[row * DI + d];
        float dtt = dt[row * DI + d];
        float bt  = bc[row * 32 + s];
        float ct  = bc[row * 32 + 16 + s];
        hst = expf(a * dtt) * hst + (dtt * xt) * bt;
        float c = hst * ct;
        c += __shfl_xor(c, 8, 16);
        c += __shfl_xor(c, 4, 16);
        c += __shfl_xor(c, 2, 16);
        c += __shfl_xor(c, 1, 16);
        if (s == 0) {
            float y = c + Dd * xt;
            float z = xz[row * (2 * DI) + DI + d];
            yz[row * DI + d] = y * (z / (1.f + expf(-z)));
        }
    }
}

// mean over L per (batch, d_model)
__global__ void mean_rows(const float* __restrict__ hn, float* __restrict__ out) {
    int b = blockIdx.x, d = threadIdx.x;
    float s = 0.f;
    for (int l = 0; l < LL; ++l)
        s += hn[((size_t)b * LL + l) * DM + d];
    out[b * DM + d] = s * (1.0f / LL);
}

// ---------------------------------------------------------------------------
extern "C" void kernel_launch(void* const* d_in, const int* in_sizes, int n_in,
                              void* d_out, int out_size, void* d_ws, size_t ws_size,
                              hipStream_t stream) {
    const float* x       = (const float*)d_in[0];
    const float* w_in    = (const float*)d_in[1];
    const float* b_in    = (const float*)d_in[2];
    const float* ln_in_g = (const float*)d_in[3];
    const float* ln_in_b = (const float*)d_in[4];
    const float* ln_g    = (const float*)d_in[5];
    const float* ln_b    = (const float*)d_in[6];
    const float* W_inpr  = (const float*)d_in[7];
    const float* conv_w  = (const float*)d_in[8];
    const float* conv_b  = (const float*)d_in[9];
    const float* W_dt    = (const float*)d_in[10];
    const float* b_dt    = (const float*)d_in[11];
    const float* W_B     = (const float*)d_in[12];
    const float* W_C     = (const float*)d_in[13];
    const float* A_log   = (const float*)d_in[14];
    const float* Dp      = (const float*)d_in[15];
    const float* W_out   = (const float*)d_in[16];
    const float* ln_f_g  = (const float*)d_in[17];
    const float* ln_f_b  = (const float*)d_in[18];

    char*  ws  = (char*)d_ws;
    size_t off = 0;
    auto alloc = [&](size_t bytes) -> void* {
        void* p = ws + off;
        off += (bytes + 255) & ~(size_t)255;
        return p;
    };
    float* h   = (float*)alloc((size_t)MROWS * DM * 4);
    float* hn  = (float*)alloc((size_t)MROWS * DM * 4);
    float* xzb = (float*)alloc((size_t)MROWS * 2 * DI * 4);
    float* xcv = (float*)alloc((size_t)MROWS * DI * 4);
    float* dtb = (float*)alloc((size_t)MROWS * DI * 4);
    float* bcb = (float*)alloc((size_t)MROWS * 32 * 4);
    float* yzb = (float*)alloc((size_t)MROWS * DI * 4);
    _Float16* pk_in  = (_Float16*)alloc((size_t)4 * DM * 2 * DI * 2);
    _Float16* pk_dt  = (_Float16*)alloc((size_t)4 * DI * DI * 2);
    _Float16* pk_bc  = (_Float16*)alloc((size_t)4 * DI * 32 * 2);
    _Float16* pk_out = (_Float16*)alloc((size_t)4 * DI * DM * 2);

    // pack all weight matrices to f16 WMMA fragments
    for (int i = 0; i < 4; ++i) {
        {   // W_inproj: 256 x 1024
            int total = DM * 2 * DI;
            pack_w_f16<<<(total + 255) / 256, 256, 0, stream>>>(
                W_inpr + (size_t)i * total, pk_in + (size_t)i * total, DM, 2 * DI);
        }
        {   // W_dt: 512 x 512
            int total = DI * DI;
            pack_w_f16<<<(total + 255) / 256, 256, 0, stream>>>(
                W_dt + (size_t)i * total, pk_dt + (size_t)i * total, DI, DI);
        }
        {   // W_B -> cols 0..15, W_C -> cols 16..31 of a K x 32 packed matrix
            int total = DI * DSTATE;    // 8192
            pack_w_f16<<<(total + 255) / 256, 256, 0, stream>>>(
                W_B + (size_t)i * total, pk_bc + (size_t)i * DI * 32, DI, DSTATE);
            pack_w_f16<<<(total + 255) / 256, 256, 0, stream>>>(
                W_C + (size_t)i * total, pk_bc + (size_t)i * DI * 32 + (size_t)DI * 16,
                DI, DSTATE);
        }
        {   // W_out: 512 x 256
            int total = DI * DM;
            pack_w_f16<<<(total + 255) / 256, 256, 0, stream>>>(
                W_out + (size_t)i * total, pk_out + (size_t)i * total, DI, DM);
        }
    }

    // stem
    input_proj_ln<<<MROWS, 256, 0, stream>>>(x, w_in, b_in, ln_in_g, ln_in_b, h);

    for (int i = 0; i < 4; ++i) {
        ln_rows<<<MROWS, 256, 0, stream>>>(h, ln_g + i * DM, ln_b + i * DM, hn);

        gemm_wmma<DM, 2 * DI, 0><<<dim3(MROWS / 128, (2 * DI) / 32), 256, 0, stream>>>(
            hn, pk_in + (size_t)i * DM * 2 * DI, xzb, nullptr);

        conv_silu<<<(MROWS * DI) / 256, 256, 0, stream>>>(
            xzb, conv_w + (size_t)i * DI * 4, conv_b + (size_t)i * DI, xcv);

        gemm_wmma<DI, DI, 1><<<dim3(MROWS / 128, DI / 32), 256, 0, stream>>>(
            xcv, pk_dt + (size_t)i * DI * DI, dtb, b_dt + (size_t)i * DI);

        gemm_wmma<DI, 32, 0><<<dim3(MROWS / 128, 1), 256, 0, stream>>>(
            xcv, pk_bc + (size_t)i * DI * 32, bcb, nullptr);

        ssm_scan<<<BB * (DI / 16), 256, 0, stream>>>(
            xcv, dtb, bcb, xzb, A_log + (size_t)i * DI * DSTATE, Dp + (size_t)i * DI, yzb);

        gemm_wmma<DI, DM, 2><<<dim3(MROWS / 128, DM / 32), 256, 0, stream>>>(
            yzb, pk_out + (size_t)i * DI * DM, h, nullptr);
    }

    // final LN + mean over sequence
    ln_rows<<<MROWS, 256, 0, stream>>>(h, ln_f_g, ln_f_b, hn);
    mean_rows<<<BB, 256, 0, stream>>>(hn, (float*)d_out);
}
